// GraphAttentionLayer_36447092474025
// MI455X (gfx1250) — matrix operand order repaired and checked by reading, not compiled, not run on hardware
//
#include <hip/hip_runtime.h>
#include <hip/hip_bf16.h>
#include <math.h>

// ---------------------------------------------------------------------------
// GAT layer, fused flash-attention style for gfx1250 (wave32, WMMA 16x16x32).
//   K1: x = fc(input)          (f16 WMMA, f32 accum) -> x_f32, xT_f16
//   K2: ax = x@w_ax+b, an = x@w_an+b
//   K3: TDM-streamed adjacency -> LDS (double buffered), fused masked online
//       softmax + (ws @ x) via f16 WMMA + residual + ELU
// ---------------------------------------------------------------------------

typedef __attribute__((ext_vector_type(16))) _Float16     v16h;
typedef __attribute__((ext_vector_type(8)))  _Float16     v8h;
typedef __attribute__((ext_vector_type(8)))  float        v8f;
typedef __attribute__((ext_vector_type(4)))  int          v4i;
typedef __attribute__((ext_vector_type(4)))  float        v4f;
typedef __attribute__((ext_vector_type(4)))  unsigned int u32x4;
typedef __attribute__((ext_vector_type(8)))  int          i32x8;
typedef __attribute__((ext_vector_type(4)))  int          i32x4;

union V16 { v16h v; v8h h[2]; };

#define N_NODES 8192
#define FIN     256
#define FOUT    128
#define NEG_BIG (-9.0e15f)
#define ADJ_PAD 8              // 8-DWORD row skew inserted by the TDM
#define NCHUNK  (N_NODES / 256)

// ===========================================================================
// Kernel 1: x = input @ W_fc + b_fc   (per block: 16 rows x 128 cols)
// ===========================================================================
__global__ __launch_bounds__(128)
void gat_fc_kernel(const float* __restrict__ input,
                   const float* __restrict__ W_fc,
                   const float* __restrict__ b_fc,
                   float* __restrict__ x_out,
                   _Float16* __restrict__ xT_out) {
  __shared__ __align__(16) _Float16 Ash[16][FIN + 8];     // input tile, f16
  __shared__ __align__(16) _Float16 WT[FOUT][FIN + 8];    // W transposed, f16

  const int tid = threadIdx.x;
  const int i0  = blockIdx.x * 16;

  {
    const int row = tid >> 3;
    const int seg = (tid & 7) * 32;
    const float4* src = (const float4*)(input + (i0 + row) * FIN + seg);
#pragma unroll
    for (int k = 0; k < 8; ++k) {
      float4 f = src[k];
      const int c = seg + k * 4;
      Ash[row][c + 0] = (_Float16)f.x;
      Ash[row][c + 1] = (_Float16)f.y;
      Ash[row][c + 2] = (_Float16)f.z;
      Ash[row][c + 3] = (_Float16)f.w;
    }
  }
  {
    const int c = tid;  // 128 threads == 128 cols
    for (int k = 0; k < FIN; ++k)
      WT[c][k] = (_Float16)W_fc[k * FOUT + c];
  }
  __syncthreads();

  const int lane = tid & 31;
  const int wave = tid >> 5;
  const int hi   = lane >> 4;
  const int ln   = lane & 15;

#pragma unroll
  for (int sub = 0; sub < 2; ++sub) {
    const int colbase = (wave * 2 + sub) * 16;
    v8f acc;
    const float bias = b_fc[colbase + ln];
#pragma unroll
    for (int v = 0; v < 8; ++v) acc[v] = bias;

#pragma unroll
    for (int kt = 0; kt < 8; ++kt) {
      const int kbase = kt * 32 + hi * 8;
      V16 A, B;
      A.h[0] = *(const v8h*)&Ash[ln][kbase];
      A.h[1] = *(const v8h*)&Ash[ln][kbase + 16];
      const int kb2 = kt * 32 + hi * 8;
      B.h[0] = *(const v8h*)&WT[colbase + ln][kb2];
      B.h[1] = *(const v8h*)&WT[colbase + ln][kb2 + 16];
      acc = __builtin_amdgcn_wmma_f32_16x16x32_f16(false, A.v, false, B.v,
                                                   (short)0, acc, false, false);
    }
#pragma unroll
    for (int v = 0; v < 8; ++v) {
      const int row = i0 + v + hi * 8;
      const int col = colbase + ln;
      x_out[row * FOUT + col]           = acc[v];
      xT_out[(long)col * N_NODES + row] = (_Float16)acc[v];
    }
  }
}

// ===========================================================================
// Kernel 2: per-row attention projections
// ===========================================================================
__global__ __launch_bounds__(256)
void gat_att_kernel(const float* __restrict__ x,
                    const float* __restrict__ w_ax, const float* __restrict__ b_ax,
                    const float* __restrict__ w_an, const float* __restrict__ b_an,
                    float* __restrict__ ax, float* __restrict__ an) {
  const int i = blockIdx.x * blockDim.x + threadIdx.x;
  if (i >= N_NODES) return;
  const float4* xr = (const float4*)(x + (long)i * FOUT);
  const float4* wa = (const float4*)w_ax;
  const float4* wn = (const float4*)w_an;
  float sa = 0.f, sn = 0.f;
#pragma unroll
  for (int k = 0; k < FOUT / 4; ++k) {
    float4 v = xr[k], a = wa[k], n = wn[k];
    sa += v.x * a.x + v.y * a.y + v.z * a.z + v.w * a.w;
    sn += v.x * n.x + v.y * n.y + v.z * n.z + v.w * n.w;
  }
  ax[i] = sa + b_ax[0];
  an[i] = sn + b_an[0];
}

// ===========================================================================
// TDM issue: DMA one 16x256 int32 adjacency tile into LDS, with an 8-DWORD
// row skew (pad_interval=256 DW, pad_amount=8 DW) for bank-conflict-free
// ds_load_b128 consumption.  D# per ISA ch.8 (2D tensor, groups 2/3 null).
// ===========================================================================
__device__ __forceinline__
void issue_adj_tdm(const int* gtile, unsigned ldsOff) {
  const unsigned long long ga = (unsigned long long)(const void*)gtile;
  const u32x4 g0 = {
      1u,                                             // count=1 (user D#)
      ldsOff,                                         // lds_addr
      (unsigned)(ga & 0xFFFFFFFFu),                   // global_addr[31:0]
      ((unsigned)((ga >> 32) & 0x01FFFFFFu)) | 0x80000000u  // [56:32] | type=2
  };
  const i32x8 g1 = {
      (int)((2u << 16) | (1u << 20) | (7u << 22) | (7u << 25)),
      //     data_size=4B | pad_en  | interval=256DW | amount=8DW
      (int)(8192u << 16),          // tensor_dim0 = 8192 (low 16 -> [31:16])
      (int)(8192u << 16),          // dim0 hi=0, tensor_dim1 = 8192 (low 16)
      (int)(256u << 16),           // dim1 hi=0, tile_dim0 = 256
      16,                          // tile_dim1 = 16, tile_dim2 = 0
      8192,                        // tensor_dim0_stride[31:0]
      0,                           // stride0 hi | dim1_stride low16 (=0)
      0x400                        // tensor_dim1_stride = 64M -> [47:16]
  };
  const i32x4 z4 = {0, 0, 0, 0};
#if defined(__clang_major__) && (__clang_major__ >= 23)
  const i32x8 z8 = {0, 0, 0, 0, 0, 0, 0, 0};
  __builtin_amdgcn_tensor_load_to_lds(g0, g1, z4, z4, z8, 0);
#else
  __builtin_amdgcn_tensor_load_to_lds(g0, g1, z4, z4, 0);
#endif
}

// ===========================================================================
// Kernel 3: fused masked online-softmax + aggregation + residual + ELU.
// One block = 16 rows; adjacency streamed by the TDM into double-buffered
// LDS; 8 waves consume disjoint 32-col strips of each 256-col chunk.
// ===========================================================================
__global__ __launch_bounds__(256)
void gat_softmax_agg_kernel(const int* __restrict__ adj,
                            const float* __restrict__ x,
                            const _Float16* __restrict__ xT,
                            const float* __restrict__ axv,
                            const float* __restrict__ anv,
                            float* __restrict__ out) {
  __shared__ __align__(16) int   adjSh[2][16][256 + ADJ_PAD];  // 2 x 16.5KB
  __shared__ __align__(16) float accSh[16][FOUT];              // 8KB
  __shared__ float mSh[8][16], lSh[8][16], lgSh[16];

  const int tid  = threadIdx.x;
  const int lane = tid & 31;
  const int wave = tid >> 5;
  const int hi   = lane >> 4;
  const int ln   = lane & 15;
  const int i0   = blockIdx.x * 16;

  for (int k = tid; k < 16 * FOUT; k += 256) ((float*)accSh)[k] = 0.f;

  const int* adjRow0 = adj + (long)i0 * N_NODES;  // block's 16-row band
  // prime the double buffer (wave 0 owns the TDM / TENSORcnt)
  if (wave == 0) {
    issue_adj_tdm(adjRow0,       (unsigned)(size_t)(void*)&adjSh[0][0][0]);
    issue_adj_tdm(adjRow0 + 256, (unsigned)(size_t)(void*)&adjSh[1][0][0]);
  }
  __syncthreads();

  const float axr = axv[i0 + ln];

  float m = -INFINITY, l = 0.f;
  v8f acc[8];
#pragma unroll
  for (int t = 0; t < 8; ++t)
#pragma unroll
    for (int v = 0; v < 8; ++v) acc[t][v] = 0.f;

  for (int it = 0; it < NCHUNK; ++it) {
    const int b     = it & 1;
    const int jbase = it * 256 + wave * 32;
    const int cl    = wave * 32 + hi * 8;     // column base inside the chunk

    if (wave == 0)
      __builtin_amdgcn_s_wait_tensorcnt((short)1);  // oldest chunk landed
    __syncthreads();                                // publish LDS to all waves

    // adjacency from LDS (row skew -> conflict-light b128 reads)
    const v4i* ap = (const v4i*)&adjSh[b][ln][cl];
    const v4i q0 = ap[0];
    const v4i q1 = ap[1];
    const v4i q2 = ap[4];      // +16 cols
    const v4i q3 = ap[5];      // +20 cols
    const v4f* np = (const v4f*)(anv + jbase + hi * 8);
    const v4f n0 = np[0], n1 = np[1], n2 = np[4], n3 = np[5];
    if (it + 1 < NCHUNK)
      __builtin_prefetch(anv + jbase + hi * 8 + 256, 0, 3);

    const int   av[16] = {q0.x, q0.y, q0.z, q0.w, q1.x, q1.y, q1.z, q1.w,
                          q2.x, q2.y, q2.z, q2.w, q3.x, q3.y, q3.z, q3.w};
    const float nv[16] = {n0.x, n0.y, n0.z, n0.w, n1.x, n1.y, n1.z, n1.w,
                          n2.x, n2.y, n2.z, n2.w, n3.x, n3.y, n3.z, n3.w};

    float s[16];
    float cmax = NEG_BIG;
#pragma unroll
    for (int e = 0; e < 16; ++e) {
      s[e] = (av[e] > 0) ? (axr + nv[e]) : NEG_BIG;
      cmax = fmaxf(cmax, s[e]);
    }
    cmax = fmaxf(cmax, __shfl_xor(cmax, 16));      // row r in lanes r, r+16
    const float mnew  = fmaxf(m, cmax);
    const float scale = __expf(m - mnew);          // first iter: exp(-inf)=0

    V16 P;
    float psum = 0.f;
#pragma unroll
    for (int e = 0; e < 16; ++e) {
      const float p = __expf(s[e] - mnew);
      psum += p;
      P.v[e] = (_Float16)p;
    }
    psum += __shfl_xor(psum, 16);
    l = l * scale + psum;
    m = mnew;

    float fac[8];
#pragma unroll
    for (int v = 0; v < 8; ++v) fac[v] = __shfl(scale, hi ? (v + 8) : v);
#pragma unroll
    for (int t = 0; t < 8; ++t)
#pragma unroll
      for (int v = 0; v < 8; ++v) acc[t][v] *= fac[v];

    // acc += P(16x32) @ x_chunk(32x128): 8 WMMAs, B from xT (L2-resident)
#pragma unroll
    for (int t = 0; t < 8; ++t) {
      const _Float16* bp = xT + (long)(t * 16 + ln) * N_NODES + jbase + hi * 8;
      V16 B;
      B.h[0] = *(const v8h*)bp;
      B.h[1] = *(const v8h*)(bp + 16);
      acc[t] = __builtin_amdgcn_wmma_f32_16x16x32_f16(false, P.v, false, B.v,
                                                      (short)0, acc[t], false, false);
    }

    __syncthreads();   // everyone done with buffer b before TDM overwrites it
    if (wave == 0 && (it + 2) < NCHUNK)
      issue_adj_tdm(adjRow0 + (it + 2) * 256,
                    (unsigned)(size_t)(void*)&adjSh[b][0][0]);
  }

  // ---- cross-wave softmax merge ----
  if (lane < 16) { mSh[wave][lane] = m; lSh[wave][lane] = l; }
  __syncthreads();

  float mg = -INFINITY;
#pragma unroll
  for (int w2 = 0; w2 < 8; ++w2) mg = fmaxf(mg, mSh[w2][ln]);
  float lg = 0.f;
#pragma unroll
  for (int w2 = 0; w2 < 8; ++w2) lg += lSh[w2][ln] * __expf(mSh[w2][ln] - mg);
  if (tid < 16) lgSh[tid] = lg;

  const float fw = __expf(m - mg);
  float fac2[8];
#pragma unroll
  for (int v = 0; v < 8; ++v) fac2[v] = __shfl(fw, hi ? (v + 8) : v);

#pragma unroll
  for (int t = 0; t < 8; ++t) {
    const int col = t * 16 + ln;
#pragma unroll
    for (int v = 0; v < 8; ++v)
      atomicAdd(&accSh[v + hi * 8][col], acc[t][v] * fac2[v]);  // ds_add_f32
  }
  __syncthreads();

  // epilogue: out = elu(x + acc / l)
  const int col = wave * 16 + ln;
#pragma unroll
  for (int i = 0; i < 8; ++i) {
    const int row = hi * 8 + i;
    const float val = accSh[row][col] / lgSh[row];
    const float o = x[(long)(i0 + row) * FOUT + col] + val;
    out[(long)(i0 + row) * FOUT + col] = (o > 0.f) ? o : (__expf(o) - 1.f);
  }
}

// ===========================================================================
extern "C" void kernel_launch(void* const* d_in, const int* in_sizes, int n_in,
                              void* d_out, int out_size, void* d_ws, size_t ws_size,
                              hipStream_t stream) {
  const float* input = (const float*)d_in[0];
  const int*   adj   = (const int*)d_in[1];
  const float* W_fc  = (const float*)d_in[2];
  const float* b_fc  = (const float*)d_in[3];
  const float* w_ax  = (const float*)d_in[4];
  const float* b_ax  = (const float*)d_in[5];
  const float* w_an  = (const float*)d_in[6];
  const float* b_an  = (const float*)d_in[7];
  float* out = (float*)d_out;

  char* ws = (char*)d_ws;
  float*    x_f32 = (float*)ws;                                      // 4 MB
  _Float16* xT    = (_Float16*)(ws + (size_t)N_NODES * FOUT * 4);    // 2 MB
  float*    axv   = (float*)(ws + (size_t)N_NODES * FOUT * 6);       // 32 KB
  float*    anv   = (float*)(ws + (size_t)N_NODES * FOUT * 6 + (size_t)N_NODES * 4);

  gat_fc_kernel<<<N_NODES / 16, 128, 0, stream>>>(input, W_fc, b_fc, x_f32, xT);
  gat_att_kernel<<<N_NODES / 256, 256, 0, stream>>>(x_f32, w_ax, b_ax, w_an, b_an, axv, anv);
  gat_softmax_agg_kernel<<<N_NODES / 16, 256, 0, stream>>>(adj, x_f32, xT, axv, anv, out);
}
